// DecoderOnlyTransformer_62740882259977
// MI455X (gfx1250) — matrix operand order repaired and checked by reading, not compiled, not run on hardware
//
#include <hip/hip_runtime.h>
#include <hip/hip_bf16.h>

// ---------------- problem constants (match reference) ----------------
constexpr int   Vc  = 32000;
constexpr int   Dc  = 1024;
constexpr int   Hc  = 16;
constexpr int   Lc  = 4;
constexpr int   DFFc= 4096;
constexpr int   Bc  = 2;
constexpr int   Sc  = 1024;
constexpr int   DKc = Dc / Hc;          // 64
constexpr float EPSc = 1e-5f;

// ---------------- WMMA types ----------------
typedef __attribute__((ext_vector_type(16))) __bf16 v16bf;
typedef __attribute__((ext_vector_type(8)))  float  v8f;

__device__ __forceinline__ unsigned short f32_to_bf16(float f) {
  unsigned int u = __float_as_uint(f);
  unsigned int r = (u + 0x7FFFu + ((u >> 16) & 1u)) >> 16;   // round-to-nearest-even
  return (unsigned short)r;
}

enum { FLAG_TRANSB = 1, FLAG_RELU = 2, FLAG_CAUSAL = 4 };

// ---------------- batched bf16-WMMA GEMM ----------------
// C[z] = scale * (A[z] x B[z]) + bias  (+mask/relu per flags)
// Block tile 128x128, BK=32; 8 waves arranged 4(M) x 2(N); each wave owns a
// 32x64 tile = 2x4 WMMA accumulators -> 8 v_wmma per wave per K-step.
// z = blockIdx.z ; outer = z / innerCount ; inner = z % innerCount ; per-matrix
// pointer offset = outer*strideOuter + inner*strideInner (element units).
#define BM_T 128
#define BN_T 128
#define BK_T 32
#define LDSS 40   // padded row stride (ushorts) -> 80B, keeps 16B alignment

__global__ __launch_bounds__(256)
void gemm_bf16_kernel(const float* __restrict__ A,  long lda, long sAo, long sAi,
                      const float* __restrict__ Bm, long ldb, long sBo, long sBi,
                      float*       __restrict__ C,  long ldc, long sCo, long sCi,
                      const float* __restrict__ bias,
                      int M, int N, int K, float scale, int flags, int innerCount) {
  __shared__ __align__(16) unsigned short As[BM_T][LDSS];
  __shared__ __align__(16) unsigned short Bs[BN_T][LDSS];   // stored transposed: Bs[n][k]

  const int tid   = threadIdx.x;
  const int wave  = tid >> 5;
  const int lane  = tid & 31;
  const int wm    = wave >> 1;        // 0..3  (M direction, 32 rows each)
  const int wn    = wave & 1;         // 0..1  (N direction, 64 cols each)
  const int laneN = lane & 15;
  const int khalf = lane >> 4;

  const int z     = blockIdx.z;
  const int outer = z / innerCount;
  const int inner = z % innerCount;
  const float* Ab = A  + (size_t)outer * sAo + (size_t)inner * sAi;
  const float* Bb = Bm + (size_t)outer * sBo + (size_t)inner * sBi;
  float*       Cb = C  + (size_t)outer * sCo + (size_t)inner * sCi;

  const int m0 = blockIdx.y * BM_T;
  const int n0 = blockIdx.x * BN_T;

  v8f acc[2][4];
#pragma unroll
  for (int i = 0; i < 2; ++i)
#pragma unroll
    for (int j = 0; j < 4; ++j)
#pragma unroll
      for (int e = 0; e < 8; ++e) acc[i][j][e] = 0.0f;

  for (int k0 = 0; k0 < K; k0 += BK_T) {
    // ---- stage A tile (128x32), f32 -> bf16 ----
    {
      const int r  = tid >> 1;            // 0..127
      const int cs = (tid & 1) << 4;      // 0,16
      const float* src = Ab + (size_t)(m0 + r) * lda + (k0 + cs);
#pragma unroll
      for (int j = 0; j < 16; ++j) As[r][cs + j] = f32_to_bf16(src[j]);
    }
    // ---- stage B tile transposed into Bs[n][k] ----
    if (flags & FLAG_TRANSB) {            // B given as [N,K] row-major
      const int n  = tid >> 1;            // 0..127
      const int cs = (tid & 1) << 4;      // 0,16
      const int ng = n0 + n;
      const float* src = Bb + (size_t)ng * ldb + (k0 + cs);
#pragma unroll
      for (int j = 0; j < 16; ++j)
        Bs[n][cs + j] = (ng < N) ? f32_to_bf16(src[j]) : (unsigned short)0;
    } else {                              // B given as [K,N] row-major
      const int k  = tid >> 3;            // 0..31
      const int ns = (tid & 7) << 4;      // 0..112
      const float* src = Bb + (size_t)(k0 + k) * ldb + (n0 + ns);
#pragma unroll
      for (int j = 0; j < 16; ++j) {
        const int ng = n0 + ns + j;
        Bs[ns + j][k] = (ng < N) ? f32_to_bf16(src[j]) : (unsigned short)0;
      }
    }
    __syncthreads();

    // ---- load WMMA fragments per ISA layout ----
    union FragU { v16bf v; unsigned int w[8]; } afr[2], bfr[4];
#pragma unroll
    for (int t = 0; t < 2; ++t) {
      // A 16x32: lane holds row M=laneN; elems 0..7 -> K = khalf*8 + 0..7,
      //          elems 8..15 -> K = 16 + khalf*8 + 0..7
      const unsigned int* pa0 =
          (const unsigned int*)&As[wm * 32 + t * 16 + laneN][khalf * 8];
      const unsigned int* pa1 =
          (const unsigned int*)&As[wm * 32 + t * 16 + laneN][16 + khalf * 8];
#pragma unroll
      for (int j = 0; j < 4; ++j) { afr[t].w[j] = pa0[j]; afr[t].w[4 + j] = pa1[j]; }
    }
#pragma unroll
    for (int t = 0; t < 4; ++t) {
      // B 32x16: lane holds column N=laneN; elems 0..7 -> K = khalf*8 + 0..7,
      //          elems 8..15 -> K = 16 + khalf*8 + 0..7
      const unsigned int* pb0 =
          (const unsigned int*)&Bs[wn * 64 + t * 16 + laneN][khalf * 8];
      const unsigned int* pb1 =
          (const unsigned int*)&Bs[wn * 64 + t * 16 + laneN][16 + khalf * 8];
#pragma unroll
      for (int j = 0; j < 4; ++j) { bfr[t].w[j] = pb0[j]; bfr[t].w[4 + j] = pb1[j]; }
    }

#pragma unroll
    for (int ti = 0; ti < 2; ++ti)
#pragma unroll
      for (int tj = 0; tj < 4; ++tj)
        acc[ti][tj] = __builtin_amdgcn_wmma_f32_16x16x32_bf16(
            false, afr[ti].v, false, bfr[tj].v,
            (short)0, acc[ti][tj], false, false);
    __syncthreads();
  }

  // ---- epilogue: scale, bias, causal mask, relu, store ----
#pragma unroll
  for (int ti = 0; ti < 2; ++ti) {
#pragma unroll
    for (int tj = 0; tj < 4; ++tj) {
      const int mBase = m0 + wm * 32 + ti * 16 + khalf * 8;   // C: M = khalf*8 + v
      const int nIdx  = n0 + wn * 64 + tj * 16 + laneN;       // C: N = laneN
      if (nIdx < N) {
        const float bv = bias ? bias[nIdx] : 0.0f;
#pragma unroll
        for (int e = 0; e < 8; ++e) {
          const int m = mBase + e;
          float val = acc[ti][tj][e] * scale + bv;
          if ((flags & FLAG_CAUSAL) && (nIdx > m)) val = -__builtin_inff();
          if (flags & FLAG_RELU) val = fmaxf(val, 0.0f);
          Cb[(size_t)m * ldc + nIdx] = val;
        }
      }
    }
  }
}

// ---------------- embedding + sinusoidal positional encoding ----------------
__global__ __launch_bounds__(256)
void embed_pe_kernel(const int* __restrict__ ids, const float* __restrict__ emb,
                     float* __restrict__ x) {
  const int row = blockIdx.x;          // b*S + s
  const int s   = row % Sc;
  const int id  = ids[row];
  const float sc = 32.0f;              // sqrt(D)=sqrt(1024)
  for (int d = threadIdx.x; d < Dc; d += 256) {
    const int i = d >> 1;
    const float div = __expf(-(float)(2 * i) * (9.210340371976184f / (float)Dc));
    const float ang = (float)s * div;
    const float pe  = (d & 1) ? __cosf(ang) : __sinf(ang);
    x[(size_t)row * Dc + d] = emb[(size_t)id * Dc + d] * sc + pe;
  }
}

// ---------------- row softmax (in place), row length Sc ----------------
__global__ __launch_bounds__(256)
void softmax_kernel(float* __restrict__ p) {
  __shared__ float red[256];
  float* row = p + (size_t)blockIdx.x * Sc;
  const int tid = threadIdx.x;

  float mx = -__builtin_inff();
  for (int j = tid; j < Sc; j += 256) mx = fmaxf(mx, row[j]);
  red[tid] = mx; __syncthreads();
  for (int off = 128; off > 0; off >>= 1) {
    if (tid < off) red[tid] = fmaxf(red[tid], red[tid + off]);
    __syncthreads();
  }
  mx = red[0]; __syncthreads();

  float sum = 0.0f;
  for (int j = tid; j < Sc; j += 256) {
    const float e = __expf(row[j] - mx);
    row[j] = e; sum += e;
  }
  red[tid] = sum; __syncthreads();
  for (int off = 128; off > 0; off >>= 1) {
    if (tid < off) red[tid] += red[tid + off];
    __syncthreads();
  }
  const float inv = 1.0f / red[0];
  for (int j = tid; j < Sc; j += 256) row[j] *= inv;
}

// ---------------- fused residual add + LayerNorm (D=1024) ----------------
__global__ __launch_bounds__(256)
void add_ln_kernel(const float* a, const float* r,
                   const float* g, const float* b, float* out) {
  __shared__ float red[256];
  const int tid = threadIdx.x;
  const size_t base = (size_t)blockIdx.x * Dc;

  float v[4]; float s = 0.0f;
#pragma unroll
  for (int j = 0; j < 4; ++j) {
    const int d = tid + j * 256;
    v[j] = a[base + d] + r[base + d];
    s += v[j];
  }
  red[tid] = s; __syncthreads();
  for (int off = 128; off > 0; off >>= 1) {
    if (tid < off) red[tid] += red[tid + off];
    __syncthreads();
  }
  const float mu = red[0] * (1.0f / (float)Dc); __syncthreads();

  float s2 = 0.0f;
#pragma unroll
  for (int j = 0; j < 4; ++j) { const float dl = v[j] - mu; s2 += dl * dl; }
  red[tid] = s2; __syncthreads();
  for (int off = 128; off > 0; off >>= 1) {
    if (tid < off) red[tid] += red[tid + off];
    __syncthreads();
  }
  const float inv = rsqrtf(red[0] * (1.0f / (float)Dc) + EPSc);
#pragma unroll
  for (int j = 0; j < 4; ++j) {
    const int d = tid + j * 256;
    out[base + d] = (v[j] - mu) * inv * g[d] + b[d];
  }
}

// ---------------- host-side GEMM launcher ----------------
static void gemm(hipStream_t st,
                 const float* A, long lda, long sAo, long sAi,
                 const float* Bm, long ldb, long sBo, long sBi,
                 float* C, long ldc, long sCo, long sCi,
                 const float* bias, int M, int N, int K,
                 float scale, int flags, int innerCount, int batch) {
  dim3 grid((N + BN_T - 1) / BN_T, (M + BM_T - 1) / BM_T, batch);
  gemm_bf16_kernel<<<grid, 256, 0, st>>>(A, lda, sAo, sAi, Bm, ldb, sBo, sBi,
                                         C, ldc, sCo, sCi, bias, M, N, K,
                                         scale, flags, innerCount);
}

extern "C" void kernel_launch(void* const* d_in, const int* in_sizes, int n_in,
                              void* d_out, int out_size, void* d_ws, size_t ws_size,
                              hipStream_t stream) {
  const int*   ids = (const int*)  d_in[0];
  const float* emb = (const float*)d_in[1];
  const float* Wq  = (const float*)d_in[2];
  const float* bq  = (const float*)d_in[3];
  const float* Wk  = (const float*)d_in[4];
  const float* bk  = (const float*)d_in[5];
  const float* Wv  = (const float*)d_in[6];
  const float* bv  = (const float*)d_in[7];
  const float* Wo  = (const float*)d_in[8];
  const float* bo  = (const float*)d_in[9];
  const float* g1  = (const float*)d_in[10];
  const float* b1  = (const float*)d_in[11];
  const float* W1  = (const float*)d_in[12];
  const float* bf1 = (const float*)d_in[13];
  const float* W2  = (const float*)d_in[14];
  const float* bf2 = (const float*)d_in[15];
  const float* g2  = (const float*)d_in[16];
  const float* b2  = (const float*)d_in[17];
  const float* Wlm = (const float*)d_in[18];

  const int BS = Bc * Sc;                       // 2048
  const size_t BSD = (size_t)BS * Dc;           // activations
  float* ws   = (float*)d_ws;
  float* x    = ws;                             // [BS, D]
  float* qb   = x  + BSD;                       // [BS, D]
  float* kb   = qb + BSD;                       // [BS, D]
  float* vb   = kb + BSD;                       // [BS, D]
  float* t0   = vb + BSD;                       // [BS, D]
  float* hbuf = t0 + BSD;                       // [BS, DFF]

  float* logits   = (float*)d_out;                      // [B,S,V]
  float* attnBase = logits + (size_t)Bc * Sc * Vc;      // [L,B,H,S,S]

  // x = emb[ids]*sqrt(D) + PE
  embed_pe_kernel<<<BS, 256, 0, stream>>>(ids, emb, x);

  const long sSD = (long)Sc * Dc;      // per-batch stride in activations
  const long sSS = (long)Sc * Sc;      // per-head stride in score maps

  for (int l = 0; l < Lc; ++l) {
    const float* Wq_l = Wq + (size_t)l * Dc * Dc;
    const float* Wk_l = Wk + (size_t)l * Dc * Dc;
    const float* Wv_l = Wv + (size_t)l * Dc * Dc;
    const float* Wo_l = Wo + (size_t)l * Dc * Dc;
    const float* W1_l = W1 + (size_t)l * Dc * DFFc;
    const float* W2_l = W2 + (size_t)l * DFFc * Dc;

    // Q,K,V projections: [BS,D] x [D,D]
    gemm(stream, x, Dc,0,0, Wq_l, Dc,0,0, qb, Dc,0,0, bq + (size_t)l*Dc, BS, Dc, Dc, 1.f, 0, 1, 1);
    gemm(stream, x, Dc,0,0, Wk_l, Dc,0,0, kb, Dc,0,0, bk + (size_t)l*Dc, BS, Dc, Dc, 1.f, 0, 1, 1);
    gemm(stream, x, Dc,0,0, Wv_l, Dc,0,0, vb, Dc,0,0, bv + (size_t)l*Dc, BS, Dc, Dc, 1.f, 0, 1, 1);

    float* attn = attnBase + (size_t)l * Bc * Hc * Sc * Sc;

    // scores = causal_mask( Q K^T / sqrt(DK) ), batched over z = b*H + h
    gemm(stream, qb, Dc, sSD, DKc,          // A offset: b*S*D + h*DK
                 kb, Dc, sSD, DKc,          // B (transposed): same head slice
                 attn, Sc, (long)Hc*sSS, sSS,
                 nullptr, Sc, Sc, DKc, 0.125f,
                 FLAG_TRANSB | FLAG_CAUSAL, Hc, Bc * Hc);

    // softmax rows in place (also final attn output for this layer)
    softmax_kernel<<<Bc * Hc * Sc, 256, 0, stream>>>(attn);

    // out = attn @ V   (write back into head-interleaved layout t0)
    gemm(stream, attn, Sc, (long)Hc*sSS, sSS,
                 vb,   Dc, sSD, DKc,
                 t0,   Dc, sSD, DKc,
                 nullptr, Sc, DKc, Sc, 1.f, 0, Hc, Bc * Hc);

    // output projection -> qb (free after scores)
    gemm(stream, t0, Dc,0,0, Wo_l, Dc,0,0, qb, Dc,0,0, bo + (size_t)l*Dc, BS, Dc, Dc, 1.f, 0, 1, 1);
    // x = LN(x + proj)
    add_ln_kernel<<<BS, 256, 0, stream>>>(x, qb, g1 + (size_t)l*Dc, b1 + (size_t)l*Dc, x);

    // FFN
    gemm(stream, x, Dc,0,0, W1_l, DFFc,0,0, hbuf, DFFc,0,0, bf1 + (size_t)l*DFFc,
         BS, DFFc, Dc, 1.f, FLAG_RELU, 1, 1);
    gemm(stream, hbuf, DFFc,0,0, W2_l, Dc,0,0, t0, Dc,0,0, bf2 + (size_t)l*Dc,
         BS, Dc, DFFc, 1.f, 0, 1, 1);
    add_ln_kernel<<<BS, 256, 0, stream>>>(x, t0, g2 + (size_t)l*Dc, b2 + (size_t)l*Dc, x);
  }

  // logits = x @ Wlm^T : [BS,D] x [V,D]^T
  gemm(stream, x, Dc,0,0, Wlm, Dc,0,0, logits, Vc,0,0, nullptr,
       BS, Vc, Dc, 1.f, FLAG_TRANSB, 1, 1);
}
